// HeteroGAE_geo_Decoder_pairwise_7112465842500
// MI455X (gfx1250) — compile-verified, matrix operand order checked by reading, not run on hardware
//
#include <hip/hip_runtime.h>
#include <hip/hip_bf16.h>
#include <math.h>

typedef __bf16 bf16_t;
typedef __attribute__((ext_vector_type(16))) __bf16 v16bf;
typedef __attribute__((ext_vector_type(8)))  float  v8f;
typedef __attribute__((ext_vector_type(4)))  unsigned int v4u;

#define BM 128
#define BN 64
#define BK 32
#define AS_STRIDE 20  /* uints per A-tile row: BK/2 + 4 pad (80B, 16B aligned) */
#define BS_STRIDE 20  /* uints per B-tile row */

__device__ inline float gelu_f(float x) {
    return 0.5f * x * (1.0f + erff(x * 0.70710678118654752f));
}
__device__ inline float sigmoid_f(float x) { return 1.0f / (1.0f + expf(-x)); }

__device__ inline void atomicMaxF(float* addr, float val) {
    if (val >= 0.0f) atomicMax((int*)addr, __float_as_int(val));
    else             atomicMin((unsigned int*)addr, __float_as_uint(val));
}

__device__ inline unsigned pack_bf16(float lo, float hi) {
    union { __bf16 h[2]; unsigned u; } p;
    p.h[0] = (__bf16)lo;
    p.h[1] = (__bf16)hi;
    return p.u;   // lowers to v_cvt_pk_bf16_f32
}

// ---------- Fragment load: 16 bf16 per lane as two 16B chunks ----------
// row_k0 points at element k0 of this lane's row (16B aligned).
// Elements [half*8 .. half*8+7] and [16+half*8 .. 16+half*8+7] per the CDNA5
// 16-bit A/B 16x16x32 VGPR layout -> exactly two b128 loads.
__device__ inline v16bf load_frag(const bf16_t* row_k0, int half) {
    union { v16bf bf; v4u u[2]; } f;
    const v4u* p = (const v4u*)row_k0;
    f.u[0] = p[half];
    f.u[1] = p[2 + half];
    return f.bf;
}

// C/D f32 16x16: VGPR r -> lanes0-15 M=r, lanes16-31 M=r+8; N=lane&15.
__device__ inline void store_c_frag(float* C, int ldc, int M, int mBase, int nBase,
                                    const v8f& acc, const float* bias, int act, int lane) {
    int half = (lane >> 4) & 1;
    int n = nBase + (lane & 15);
    float bv = bias[n];
#pragma unroll
    for (int r = 0; r < 8; ++r) {
        int m = mBase + r + half * 8;
        if (m < M) {
            float x = acc[r] + bv;
            if (act == 1) x = gelu_f(x);
            C[(size_t)m * ldc + n] = x;
        }
    }
}

// ---------- One-time weight pre-pack: W[K][N] fp32 -> Wpk[n][k/2] bf16x2 ----------
__global__ void pack_weight(const float* __restrict__ W, int K, int Nc, unsigned* __restrict__ out) {
    int i = blockIdx.x * blockDim.x + threadIdx.x;
    int Kh = K >> 1;
    if (i >= Nc * Kh) return;
    int kh = i / Nc, n = i % Nc;  // coalesced reads of rows 2kh, 2kh+1
    out[(size_t)n * Kh + kh] = pack_bf16(W[(size_t)(2 * kh) * Nc + n],
                                         W[(size_t)(2 * kh + 1) * Nc + n]);
}

// ---------- Generic bf16 WMMA GEMM, double-buffered LDS ----------
// C = act(A[MxK] * B[KxN] + bias); B pre-packed transposed. K%32==0, N%64==0.
__global__ __launch_bounds__(256) void gemm_bias_act(const float* __restrict__ A,
                                                     const unsigned* __restrict__ Bpk,
                                                     const float* __restrict__ bias,
                                                     float* __restrict__ C,
                                                     int M, int Nc, int K, int act) {
    __shared__ __align__(16) unsigned As_u[2][BM * AS_STRIDE];  // 2 x 10KB
    __shared__ __align__(16) unsigned Bs_u[2][BN * BS_STRIDE];  // 2 x 5KB
    int tid = threadIdx.x, lane = tid & 31, wave = tid >> 5;
    int half = (lane >> 4) & 1;
    int wm = (wave >> 1) * 32;  // 0,32,64,96
    int wn = (wave & 1) * 32;   // 0,32
    int m0 = blockIdx.x * BM;
    int n0 = blockIdx.y * BN;
    int Kh = K >> 1;
    int nk = K / BK;

    v8f acc[2][2];
#pragma unroll
    for (int i = 0; i < 2; ++i)
#pragma unroll
        for (int j = 0; j < 2; ++j)
#pragma unroll
            for (int r = 0; r < 8; ++r) acc[i][j][r] = 0.0f;

    auto stage = [&](int kt, int buf) {
        // A tile 128x32: 1024 float4 / 256 thr = 4; pack pairs -> uint2 LDS stores.
#pragma unroll
        for (int it = 0; it < 4; ++it) {
            int idx = tid + it * 256;
            int r = idx >> 3, c4 = idx & 7;
            int gm = m0 + r;
            if (gm >= M) gm = M - 1;  // padded rows: garbage, never stored
            const float4 f = *(const float4*)(A + (size_t)gm * K + kt * BK + c4 * 4);
            uint2 pk;
            pk.x = pack_bf16(f.x, f.y);
            pk.y = pack_bf16(f.z, f.w);
            *(uint2*)&As_u[buf][r * AS_STRIDE + c4 * 2] = pk;
        }
        // B tile 64x(32/2 uints): straight packed copy, coalesced along k.
#pragma unroll
        for (int it = 0; it < 2; ++it) {
            int idx = tid + it * 256;
            int n = idx >> 3, c2 = idx & 7;
            *(uint2*)&Bs_u[buf][n * BS_STRIDE + c2 * 2] =
                *(const uint2*)&Bpk[(size_t)(n0 + n) * Kh + kt * (BK / 2) + c2 * 2];
        }
        if ((kt + 1) * BK < K) {
            int gm = m0 + (tid >> 1);
            if (gm >= M) gm = M - 1;
            __builtin_prefetch(&A[(size_t)gm * K + (kt + 1) * BK], 0, 0);
        }
    };

    stage(0, 0);
    int buf = 0;
    for (int kt = 0; kt < nk; ++kt) {
        __syncthreads();
        if (kt + 1 < nk) stage(kt + 1, buf ^ 1);
        const bf16_t* Ab = (const bf16_t*)&As_u[buf][0];
        const bf16_t* Bb = (const bf16_t*)&Bs_u[buf][0];
        v16bf af0 = load_frag(Ab + (size_t)(wm + (lane & 15)) * (AS_STRIDE * 2), half);
        v16bf af1 = load_frag(Ab + (size_t)(wm + 16 + (lane & 15)) * (AS_STRIDE * 2), half);
        v16bf bf0 = load_frag(Bb + (size_t)(wn + (lane & 15)) * (BS_STRIDE * 2), half);
        v16bf bf1 = load_frag(Bb + (size_t)(wn + 16 + (lane & 15)) * (BS_STRIDE * 2), half);
        acc[0][0] = __builtin_amdgcn_wmma_f32_16x16x32_bf16(false, af0, false, bf0, (short)0, acc[0][0], false, false);
        acc[0][1] = __builtin_amdgcn_wmma_f32_16x16x32_bf16(false, af0, false, bf1, (short)0, acc[0][1], false, false);
        acc[1][0] = __builtin_amdgcn_wmma_f32_16x16x32_bf16(false, af1, false, bf0, (short)0, acc[1][0], false, false);
        acc[1][1] = __builtin_amdgcn_wmma_f32_16x16x32_bf16(false, af1, false, bf1, (short)0, acc[1][1], false, false);
        buf ^= 1;
    }
#pragma unroll
    for (int i = 0; i < 2; ++i)
#pragma unroll
        for (int j = 0; j < 2; ++j)
            store_c_frag(C, Nc, M, m0 + wm + i * 16, n0 + wn + j * 16, acc[i][j], bias, act, lane);
}

// ---------- Fused contact decoder: 64 pairs/block, weights pre-packed ----------
__global__ __launch_bounds__(256) void contact_decoder(const float* __restrict__ z,
                                                       const int* __restrict__ p0,
                                                       const int* __restrict__ p1,
                                                       const unsigned* __restrict__ wc1pk,
                                                       const float* __restrict__ bc1,
                                                       const unsigned* __restrict__ wc2pk,
                                                       const float* __restrict__ bc2,
                                                       const float* __restrict__ Wc3,
                                                       const float* __restrict__ bc3,
                                                       int P, float* __restrict__ out) {
    extern __shared__ __align__(16) char smem[];
    const int ZZ_SU = 256 + 4;       // uints per zz row (512 bf16 + pad)
    const int H1_S = 256 + 8;        // bf16 per h1 row (528B, 16B aligned)
    const int H2_S = 128 + 8;        // bf16 per h2 row (272B, 16B aligned)
    unsigned* zz_u = (unsigned*)smem;           // 64*260*4  = 66560 B
    bf16_t* h1 = (bf16_t*)(zz_u + 64 * ZZ_SU);  // 64*264*2  = 33792 B
    bf16_t* h2 = h1 + 64 * H1_S;                // 64*136*2  = 17408 B

    int tid = threadIdx.x, lane = tid & 31, wave = tid >> 5;
    int half = (lane >> 4) & 1;
    int base = blockIdx.x * 64;

    // Gather zz = [z[p0] | z[p1]] (z is L2-resident): float4 loads, packed b64 LDS stores.
#pragma unroll
    for (int it = 0; it < 32; ++it) {
        int idx = tid + it * 256;
        int r = idx >> 7, c4 = idx & 127;
        int pr = base + r;
        if (pr >= P) pr = P - 1;  // padded rows: values never stored
        int node = (c4 < 64) ? p0[pr] : p1[pr];
        const float4 f = *(const float4*)(z + (size_t)node * 256 + (c4 & 63) * 4);
        uint2 pk;
        pk.x = pack_bf16(f.x, f.y);
        pk.y = pack_bf16(f.z, f.w);
        *(uint2*)&zz_u[r * ZZ_SU + c4 * 2] = pk;
    }
    __syncthreads();

    const bf16_t* zzb = (const bf16_t*)zz_u;
    const bf16_t* w1b = (const bf16_t*)wc1pk;  // row n: 512 bf16 (K=512)
    const bf16_t* w2b = (const bf16_t*)wc2pk;  // row n: 256 bf16 (K=256)

    // GEMM1: h1(64x256) = gelu(zz(64x512) @ Wc1 + bc1)
    {
        int wm = (wave & 3) * 16;
        int nBase = (wave >> 2) * 128;
        for (int nt = 0; nt < 8; ++nt) {
            int n0 = nBase + nt * 16;
            int nl = n0 + (lane & 15);
            v8f acc;
#pragma unroll
            for (int r = 0; r < 8; ++r) acc[r] = 0.0f;
#pragma unroll
            for (int k0 = 0; k0 < 512; k0 += 32) {
                v16bf a = load_frag(zzb + (size_t)(wm + (lane & 15)) * (ZZ_SU * 2) + k0, half);
                v16bf b = load_frag(w1b + (size_t)nl * 512 + k0, half);
                acc = __builtin_amdgcn_wmma_f32_16x16x32_bf16(false, a, false, b, (short)0, acc, false, false);
            }
            float bv = bc1[nl];
#pragma unroll
            for (int r = 0; r < 8; ++r)
                h1[(wm + r + half * 8) * H1_S + nl] = (bf16_t)gelu_f(acc[r] + bv);
        }
    }
    __syncthreads();

    // GEMM2: h2(64x128) = gelu(h1(64x256) @ Wc2 + bc2)
    {
        int wm = (wave & 3) * 16;
        int nBase = (wave >> 2) * 64;
        for (int nt = 0; nt < 4; ++nt) {
            int n0 = nBase + nt * 16;
            int nl = n0 + (lane & 15);
            v8f acc;
#pragma unroll
            for (int r = 0; r < 8; ++r) acc[r] = 0.0f;
#pragma unroll
            for (int k0 = 0; k0 < 256; k0 += 32) {
                v16bf a = load_frag(h1 + (size_t)(wm + (lane & 15)) * H1_S + k0, half);
                v16bf b = load_frag(w2b + (size_t)nl * 256 + k0, half);
                acc = __builtin_amdgcn_wmma_f32_16x16x32_bf16(false, a, false, b, (short)0, acc, false, false);
            }
            float bv = bc2[nl];
#pragma unroll
            for (int r = 0; r < 8; ++r)
                h2[(wm + r + half * 8) * H2_S + nl] = (bf16_t)gelu_f(acc[r] + bv);
        }
    }
    __syncthreads();

    // GEMM3: 128 -> 1 dot + sigmoid.
    if (tid < 64) {
        const bf16_t* row = h2 + tid * H2_S;
        float acc = bc3[0];
#pragma unroll 8
        for (int c = 0; c < 128; ++c) acc += (float)row[c] * Wc3[c];
        int pr = base + tid;
        if (pr < P) out[pr] = sigmoid_f(acc);
    }
}

// ---------- Elementwise / graph kernels ----------
__global__ void fill_f32(float* p, float v, int n) {
    int i = blockIdx.x * blockDim.x + threadIdx.x;
    if (i < n) p[i] = v;
}

__global__ void edge_logits(const float* __restrict__ q, const float* __restrict__ k,
                            const int* __restrict__ src, const int* __restrict__ dst,
                            float* __restrict__ logits, float* __restrict__ mx, int E) {
    int i = blockIdx.x * blockDim.x + threadIdx.x;
    if (i >= E * 4) return;
    int e = i >> 2, h = i & 3;
    int d = dst[e], s = src[e];
    const float4* qr = (const float4*)(q + (size_t)d * 512 + h * 128);
    const float4* kr = (const float4*)(k + (size_t)s * 512 + h * 128);
    float acc = 0.0f;
#pragma unroll 8
    for (int t = 0; t < 32; ++t) {
        float4 a = qr[t], b = kr[t];
        acc += a.x * b.x + a.y * b.y + a.z * b.z + a.w * b.w;
    }
    acc *= 0.08838834764831845f;  // 1/sqrt(128)
    logits[i] = acc;
    atomicMaxF(&mx[d * 4 + h], acc);
}

__global__ void edge_exp(const int* __restrict__ dst, float* __restrict__ logits,
                         const float* __restrict__ mx, float* __restrict__ den, int E) {
    int i = blockIdx.x * blockDim.x + threadIdx.x;
    if (i >= E * 4) return;
    int e = i >> 2, h = i & 3;
    int d = dst[e];
    float ex = expf(logits[i] - mx[d * 4 + h]);
    logits[i] = ex;
    atomicAdd(&den[d * 4 + h], ex);
}

__global__ __launch_bounds__(256) void edge_msg(const int* __restrict__ src, const int* __restrict__ dst,
                                                const float* __restrict__ ex, const float* __restrict__ v,
                                                float* __restrict__ msg, int E) {
    int e = blockIdx.x;
    if (e >= E) return;
    int d = dst[e], s = src[e];
#pragma unroll
    for (int t = threadIdx.x; t < 512; t += 256) {
        int h = t >> 7;
        atomicAdd(&msg[(size_t)d * 512 + t], ex[e * 4 + h] * v[(size_t)s * 512 + t]);
    }
}

__global__ __launch_bounds__(128) void node_update(const float* __restrict__ msg, const float* __restrict__ den,
                                                   const float* __restrict__ s, float* __restrict__ hnew, int N) {
    int n = blockIdx.x, c = threadIdx.x;
    if (n >= N) return;
    float acc = 0.0f;
#pragma unroll
    for (int h = 0; h < 4; ++h)
        acc += msg[(size_t)n * 512 + h * 128 + c] / (den[n * 4 + h] + 1e-16f);
    float x = acc * 0.25f + s[(size_t)n * 128 + c];
    hnew[(size_t)n * 128 + c] = gelu_f(x);
}

__global__ void colsum(const float* __restrict__ x, int N, int C, int rowsPerBlock, float* __restrict__ out) {
    int c = threadIdx.x;  // blockDim == C
    int r0 = blockIdx.x * rowsPerBlock;
    int r1 = r0 + rowsPerBlock; if (r1 > N) r1 = N;
    float acc = 0.0f;
    for (int r = r0; r < r1; ++r) acc += x[(size_t)r * C + c];
    atomicAdd(&out[c], acc);
}

__global__ void gn_stats(const float* __restrict__ x, const float* __restrict__ sum,
                         const float* __restrict__ ms, int N, int C, int rowsPerBlock,
                         float* __restrict__ varsum) {
    int c = threadIdx.x;
    float sub = ms[c] * (sum[c] / (float)N);
    int r0 = blockIdx.x * rowsPerBlock;
    int r1 = r0 + rowsPerBlock; if (r1 > N) r1 = N;
    float acc = 0.0f;
    for (int r = r0; r < r1; ++r) {
        float hc = x[(size_t)r * C + c] - sub;
        acc += hc * hc;
    }
    atomicAdd(&varsum[c], acc);
}

__global__ void gn_apply(const float* __restrict__ x, const float* __restrict__ prev,
                         const float* __restrict__ sum, const float* __restrict__ varsum,
                         const float* __restrict__ gw, const float* __restrict__ gb,
                         const float* __restrict__ ms, int N, int addprev,
                         float* __restrict__ hout, float* __restrict__ zcat, int zoff) {
    int i = blockIdx.x * blockDim.x + threadIdx.x;
    if (i >= N * 128) return;
    int n = i >> 7, c = i & 127;
    float mean = sum[c] / (float)N;
    float var = varsum[c] / (float)N;
    float hc = x[i] - ms[c] * mean;
    float y = gw[c] * hc * rsqrtf(var + 1e-5f) + gb[c];
    if (addprev) y += prev[i];
    hout[i] = y;
    zcat[(size_t)n * 384 + zoff + c] = y;
}

__global__ void dyt_kernel(float* __restrict__ z, const float* __restrict__ a,
                           const float* __restrict__ w, const float* __restrict__ b, int N) {
    int i = blockIdx.x * blockDim.x + threadIdx.x;
    if (i >= N * 384) return;
    int c = i % 384;
    z[i] = w[c] * tanhf(a[0] * z[i]) + b[c];
}

__global__ __launch_bounds__(256) void resnorm(const float* __restrict__ z3, const float* __restrict__ xres,
                                               float* __restrict__ outz, float* __restrict__ zsum, int N) {
    int n = blockIdx.x, c = threadIdx.x;
    __shared__ float red[256];
    float v = z3[(size_t)n * 256 + c] + xres[(size_t)n * 256 + c];
    red[c] = v * v;
    __syncthreads();
    for (int s = 128; s > 0; s >>= 1) {
        if (c < s) red[c] += red[c + s];
        __syncthreads();
    }
    float zn = v / (sqrtf(red[0]) + 1e-10f);
    outz[(size_t)n * 256 + c] = zn;
    atomicAdd(&zsum[c], zn);
}

__global__ __launch_bounds__(128) void jv_kernel(const float* __restrict__ zsum,
                                                 const float* __restrict__ Wj1, const float* __restrict__ bj1,
                                                 const float* __restrict__ Wj2, const float* __restrict__ bj2,
                                                 int N, float* __restrict__ jv_out) {
    __shared__ float zagg[256];
    __shared__ float t1[128];
    __shared__ float red[128];
    int t = threadIdx.x;
    zagg[t]       = zsum[t] / (float)N;
    zagg[t + 128] = zsum[t + 128] / (float)N;
    __syncthreads();
    float acc = bj1[t];
    for (int c = 0; c < 256; ++c) acc += zagg[c] * Wj1[c * 128 + t];
    t1[t] = gelu_f(acc);
    __syncthreads();
    float acc2 = bj2[t];
    for (int c = 0; c < 128; ++c) acc2 += t1[c] * Wj2[c * 128 + t];
    red[t] = acc2 * acc2;
    __syncthreads();
    for (int s = 64; s > 0; s >>= 1) {
        if (t < s) red[t] += red[t + s];
        __syncthreads();
    }
    jv_out[t] = acc2 / (sqrtf(red[0]) + 1e-10f);
}

// ---------- Host orchestration ----------
static inline int ceil_div(int a, int b) { return (a + b - 1) / b; }

extern "C" void kernel_launch(void* const* d_in, const int* in_sizes, int n_in,
                              void* d_out, int out_size, void* d_ws, size_t ws_size,
                              hipStream_t stream) {
    const float* x_res = (const float*)d_in[0];
    const int* esrc = (const int*)d_in[1];
    const int* edst = (const int*)d_in[2];
    const int* cp   = (const int*)d_in[3];

    const int N = in_sizes[0] / 256;
    const int E = in_sizes[1];
    const int P = in_sizes[3] / 2;

    auto PRM = [&](int i) { return (const float*)d_in[i]; };
    const int PB = 4;  // per-layer base PB+11*l: Wq,bq,Wk,bk,Wv,bv,Ws,bs,gn_w,gn_b,gn_ms
    const float* dyt_a = PRM(PB + 33 + 0);
    const float* dyt_w = PRM(PB + 33 + 1);
    const float* dyt_b = PRM(PB + 33 + 2);
    const float* W1 = PRM(PB + 33 + 3);  const float* b1 = PRM(PB + 33 + 4);
    const float* W2 = PRM(PB + 33 + 5);  const float* b2 = PRM(PB + 33 + 6);
    const float* W3 = PRM(PB + 33 + 7);  const float* b3 = PRM(PB + 33 + 8);
    const float* Wc1 = PRM(PB + 33 + 9);  const float* bc1 = PRM(PB + 33 + 10);
    const float* Wc2 = PRM(PB + 33 + 11); const float* bc2 = PRM(PB + 33 + 12);
    const float* Wc3 = PRM(PB + 33 + 13); const float* bc3 = PRM(PB + 33 + 14);
    const float* Wj1 = PRM(PB + 33 + 15); const float* bj1 = PRM(PB + 33 + 16);
    const float* Wj2 = PRM(PB + 33 + 17); const float* bj2 = PRM(PB + 33 + 18);

    float* W = (float*)d_ws;
    size_t off = 0;
    auto alloc = [&](size_t n) { float* p = W + off; off += n; return p; };
    float* hbuf = alloc((size_t)N * 128);
    float* hnew = alloc((size_t)N * 128);
    float* qb   = alloc((size_t)N * 512);
    float* kb   = alloc((size_t)N * 512);
    float* vb   = alloc((size_t)N * 512);
    float* sb   = alloc((size_t)N * 128);
    float* lg   = alloc((size_t)E * 4);
    float* mb   = alloc((size_t)N * 4);
    float* den  = alloc((size_t)N * 4);
    float* msg  = alloc((size_t)N * 512);
    float* zcat = alloc((size_t)N * 384);
    float* z1   = alloc((size_t)N * 512);
    float* z2   = alloc((size_t)N * 512);
    float* z3   = alloc((size_t)N * 256);
    float* cs   = alloc(512);
    float* vs   = alloc(512);
    float* zsum = alloc(256);
    unsigned* wpk   = (unsigned*)alloc(131072);  // packed-weight scratch (max 512x512/2)
    unsigned* wc1pk = (unsigned*)alloc(65536);   // 256 x 256 uints
    unsigned* wc2pk = (unsigned*)alloc(16384);   // 128 x 128 uints
    (void)ws_size; (void)n_in; (void)out_size;

    float* outc  = (float*)d_out;
    float* outz  = outc + P;
    float* outjv = outz + (size_t)N * 256;

    auto gemm = [&](const float* A, const float* Bw, const float* bias, float* Cc,
                    int M, int Nc, int K, int act) {
        int tot = Nc * (K >> 1);
        pack_weight<<<ceil_div(tot, 256), 256, 0, stream>>>(Bw, K, Nc, wpk);
        dim3 grid(ceil_div(M, BM), Nc / BN);
        gemm_bias_act<<<grid, 256, 0, stream>>>(A, wpk, bias, Cc, M, Nc, K, act);
    };

    const int ROWS = 256;
    for (int l = 0; l < 3; ++l) {
        const float* A = (l == 0) ? x_res : hbuf;
        int K = (l == 0) ? 256 : 128;
        int pb = PB + 11 * l;
        gemm(A, PRM(pb + 0), PRM(pb + 1), qb, N, 512, K, 0);
        gemm(A, PRM(pb + 2), PRM(pb + 3), kb, N, 512, K, 0);
        gemm(A, PRM(pb + 4), PRM(pb + 5), vb, N, 512, K, 0);
        gemm(A, PRM(pb + 6), PRM(pb + 7), sb, N, 128, K, 0);

        fill_f32<<<ceil_div(N * 4, 256), 256, 0, stream>>>(mb, -INFINITY, N * 4);
        fill_f32<<<ceil_div(N * 4, 256), 256, 0, stream>>>(den, 0.0f, N * 4);
        fill_f32<<<ceil_div(N * 512, 256), 256, 0, stream>>>(msg, 0.0f, N * 512);

        edge_logits<<<ceil_div(E * 4, 256), 256, 0, stream>>>(qb, kb, esrc, edst, lg, mb, E);
        edge_exp<<<ceil_div(E * 4, 256), 256, 0, stream>>>(edst, lg, mb, den, E);
        edge_msg<<<E, 256, 0, stream>>>(esrc, edst, lg, vb, msg, E);
        node_update<<<N, 128, 0, stream>>>(msg, den, sb, hnew, N);

        fill_f32<<<1, 256, 0, stream>>>(cs, 0.0f, 128);
        fill_f32<<<1, 256, 0, stream>>>(vs, 0.0f, 128);
        colsum<<<ceil_div(N, ROWS), 128, 0, stream>>>(hnew, N, 128, ROWS, cs);
        gn_stats<<<ceil_div(N, ROWS), 128, 0, stream>>>(hnew, cs, PRM(pb + 10), N, 128, ROWS, vs);
        gn_apply<<<ceil_div(N * 128, 256), 256, 0, stream>>>(hnew, hbuf, cs, vs, PRM(pb + 8),
                                                             PRM(pb + 9), PRM(pb + 10), N,
                                                             (l > 0) ? 1 : 0, hbuf, zcat, l * 128);
    }

    dyt_kernel<<<ceil_div(N * 384, 256), 256, 0, stream>>>(zcat, dyt_a, dyt_w, dyt_b, N);
    gemm(zcat, W1, b1, z1, N, 512, 384, 1);
    gemm(z1, W2, b2, z2, N, 512, 512, 1);
    gemm(z2, W3, b3, z3, N, 256, 512, 0);

    fill_f32<<<1, 256, 0, stream>>>(zsum, 0.0f, 256);
    resnorm<<<N, 256, 0, stream>>>(z3, x_res, outz, zsum, N);
    jv_kernel<<<1, 128, 0, stream>>>(zsum, Wj1, bj1, Wj2, bj2, N, outjv);

    // Pack decoder weights once, then fused WMMA decoder.
    pack_weight<<<ceil_div(256 * 256, 256), 256, 0, stream>>>(Wc1, 512, 256, wc1pk);
    pack_weight<<<ceil_div(128 * 128, 256), 256, 0, stream>>>(Wc2, 256, 128, wc2pk);
    size_t smem = (size_t)(64 * (256 + 4)) * 4 + (size_t)(64 * (256 + 8)) * 2 + (size_t)(64 * (128 + 8)) * 2;
    contact_decoder<<<ceil_div(P, 64), 256, smem, stream>>>(outz, cp, cp + P, wc1pk, bc1,
                                                            wc2pk, bc2, Wc3, bc3, P, outc);
}
